// SpatioTemporalBlock_62234076119733
// MI455X (gfx1250) — matrix operand order repaired and checked by reading, not compiled, not run on hardware
//
#include <hip/hip_runtime.h>

#define BB 8
#define TT 16
#define NN 256
#define FIN 128
#define HH 128
#define HEADS 2
#define KTOP 15
#define BT (BB*TT)        /* 128  */
#define ROWS_X (BT*NN)    /* 32768 */
#define SEQ (BB*NN*3)     /* 6144 */

typedef __attribute__((ext_vector_type(16))) __bf16 v16bf;
typedef __attribute__((ext_vector_type(8)))  float  v8f;

union Frag { v16bf v; uint4 u[2]; };

__device__ __forceinline__ unsigned short f2bf(float f) {
  union { float f; unsigned u; } x; x.f = f;
  unsigned r = (x.u + 0x7FFFu + ((x.u >> 16) & 1u)) >> 16;
  return (unsigned short)r;
}

__device__ __forceinline__ v8f vzero8() {
  v8f z = {0.f,0.f,0.f,0.f,0.f,0.f,0.f,0.f}; return z;
}

// A fragment (16x32 bf16, M x K): lane r holds row m0+r.
// hi=0: K[kb..kb+7] then K[kb+16..kb+23]; hi=1: K[kb+8..15], K[kb+24..31].
__device__ __forceinline__ Frag load_a(const unsigned short* row, int kb, int hi) {
  Frag f;
  f.u[0] = *(const uint4*)(row + kb + 8*hi);
  f.u[1] = *(const uint4*)(row + kb + 16 + 8*hi);
  return f;
}
// B fragment (32x16 bf16, K x N) from row-major (N x K) weights:
// lane r holds col n0+r; hi=0: K[kb..kb+7], K[kb+16..kb+23]; hi=1: K[kb+8..15], K[kb+24..31].
__device__ __forceinline__ Frag load_b(const unsigned short* row, int kb, int hi) {
  Frag f;
  f.u[0] = *(const uint4*)(row + kb + 8*hi);
  f.u[1] = *(const uint4*)(row + kb + 16 + 8*hi);
  return f;
}
__device__ __forceinline__ v8f wmma_bf16(Frag a, Frag b, v8f c) {
  return __builtin_amdgcn_wmma_f32_16x16x32_bf16(false, a.v, false, b.v,
                                                 (short)0, c, false, false);
}

// ---------------------------------------------------------------- cvt
__global__ void cvt_bf16(const float* __restrict__ src,
                         unsigned short* __restrict__ dst, int n) {
  int i = blockIdx.x * blockDim.x + threadIdx.x;
  if (i < n) dst[i] = f2bf(src[i]);
}

// ------------------------------------------------- Wh = x @ W_gat^T
// Register-blocked: one wave computes a 16(m) x 64(n) block; the A
// fragment is loaded once per k-step and reused across 4 WMMAs.
// X: [32768 x 128] bf16, Wg: [256 x 128] bf16
// Whf : f32  [(bt*2+head)*256 + node][128]
// WhT : bf16 [(bt*2+head)*128 + f][256]   (B operand for att@Wh)
__global__ void wh_gemm(const unsigned short* __restrict__ X,
                        const unsigned short* __restrict__ Wg,
                        float* __restrict__ Whf,
                        unsigned short* __restrict__ WhT) {
  int gw   = (blockIdx.x * blockDim.x + threadIdx.x) >> 5;  // 8192 waves
  int lane = threadIdx.x & 31;
  int mt = gw >> 2;          // 2048 m-tiles
  int ng = gw & 3;           // 4 groups of 64 columns
  int r = lane & 15, hi = lane >> 4;
  const unsigned short* arow  = X  + (size_t)(mt*16 + r) * FIN;
  const unsigned short* brow0 = Wg + (size_t)(ng*64 + r) * FIN;
  v8f acc[4] = {vzero8(), vzero8(), vzero8(), vzero8()};
#pragma unroll
  for (int kb = 0; kb < FIN; kb += 32) {
    Frag a = load_a(arow, kb, hi);
#pragma unroll
    for (int q = 0; q < 4; ++q)
      acc[q] = wmma_bf16(a, load_b(brow0 + (size_t)q*16*FIN, kb, hi), acc[q]);
  }
#pragma unroll
  for (int q = 0; q < 4; ++q) {
    int ncol = ng*64 + q*16 + r;
    int head = ncol >> 7, f = ncol & 127;
#pragma unroll
    for (int v = 0; v < 8; ++v) {
      int row  = mt*16 + v + 8*hi;
      int bt   = row >> 8, node = row & 255;
      float val = acc[q][v];
      Whf[(size_t)((bt*HEADS + head)*NN + node)*HH + f] = val;
      WhT[(size_t)((bt*HEADS + head)*HH + f)*NN + node] = f2bf(val);
    }
  }
}

// ------------------------------------- per-node projections of Wh
__global__ void row_dots(const float* __restrict__ Whf,
                         const float* __restrict__ a_attn,   // [2][256]
                         const float* __restrict__ w_proj,   // [128]
                         const float* __restrict__ w_sim,    // [128]
                         float* __restrict__ s1, float* __restrict__ s2,
                         float* __restrict__ sp, float* __restrict__ ss) {
  int gw   = (blockIdx.x * blockDim.x + threadIdx.x) >> 5;   // 65536 rows
  int lane = threadIdx.x & 31;
  int head = (gw >> 8) & 1;
  const float* wh = Whf + (size_t)gw * HH;
  float p1 = 0.f, p2 = 0.f, pp = 0.f, ps = 0.f;
  for (int k = lane; k < HH; k += 32) {
    float v = wh[k];
    p1 += v * a_attn[head*256 + k];
    p2 += v * a_attn[head*256 + 128 + k];
    pp += v * w_proj[k];
    ps += v * w_sim[k];
  }
  for (int o = 16; o; o >>= 1) {
    p1 += __shfl_xor(p1, o); p2 += __shfl_xor(p2, o);
    pp += __shfl_xor(pp, o); ps += __shfl_xor(ps, o);
  }
  if (lane == 0) { s1[gw] = p1; s2[gw] = p2; sp[gw] = pp; ss[gw] = ps; }
}

// --------------------------- attention scores + top/bottom-k softmax
// one wave per row (bt, head, i); 4 waves per 128-thread block
__global__ void attn_topk(const float* __restrict__ s1, const float* __restrict__ s2,
                          const float* __restrict__ sp, const float* __restrict__ ss,
                          const float* __restrict__ proj, const float* __restrict__ sim,
                          const int* __restrict__ quad,  const float* __restrict__ qb,
                          unsigned short* __restrict__ Att) {
  __shared__ float ebuf[4][256];
  int gw   = (blockIdx.x * blockDim.x + threadIdx.x) >> 5;   // 65536 rows
  int lane = threadIdx.x & 31;
  int wl   = (threadIdx.x >> 5) & 3;
  int i    = gw & 255;
  int head = (gw >> 8) & 1;
  int bt   = gw >> 9;
  int base = (bt*HEADS + head)*NN;
  float s1i = s1[base + i];
  const float* projrow = proj + (size_t)(bt*NN + i)*NN;
  const float* simrow  = sim  + (size_t)(bt*NN + i)*NN;
  const int*   qrow    = quad + (size_t)(bt*NN + i)*NN*2;
  float ev[8];
#pragma unroll
  for (int jj = 0; jj < 8; ++jj) {
    int j = lane + jj*32;
    float e = s1i + s2[base + j];
    e = (e >= 0.f) ? e : 0.2f*e;                 // leaky relu
    e += projrow[j]*sp[base + j] + simrow[j]*ss[base + j];
    e += qb[qrow[2*j]*4 + qrow[2*j + 1]];
    ev[jj] = e;
    ebuf[wl][j] = e;
  }
  __syncthreads();
  int cg[8], cl[8];
#pragma unroll
  for (int jj = 0; jj < 8; ++jj) { cg[jj] = 0; cl[jj] = 0; }
  for (int l = 0; l < 256; ++l) {
    float o = ebuf[wl][l];
#pragma unroll
    for (int jj = 0; jj < 8; ++jj) {
      cg[jj] += (o > ev[jj]) ? 1 : 0;
      cl[jj] += (o < ev[jj]) ? 1 : 0;
    }
  }
  float mval[8];
  float mv = -1e30f;
#pragma unroll
  for (int jj = 0; jj < 8; ++jj) {
    bool keep = (cg[jj] < KTOP) || (cl[jj] < KTOP);
    float m_ = keep ? ev[jj] : -1.0e9f;
    mval[jj] = m_;
    mv = fmaxf(mv, m_);
  }
  for (int o = 16; o; o >>= 1) mv = fmaxf(mv, __shfl_xor(mv, o));
  float ex[8]; float sum = 0.f;
#pragma unroll
  for (int jj = 0; jj < 8; ++jj) { ex[jj] = __expf(mval[jj] - mv); sum += ex[jj]; }
  for (int o = 16; o; o >>= 1) sum += __shfl_xor(sum, o);
  float inv = 1.f / sum;
#pragma unroll
  for (int jj = 0; jj < 8; ++jj)
    Att[(size_t)gw*NN + lane + jj*32] = f2bf(ex[jj]*inv);
}

// ------------------------------------------------- hp = att @ Wh
// Register-blocked: one wave computes 16(i) x 64(f); A frag reused x4.
__global__ void hp_gemm(const unsigned short* __restrict__ Att,  // [(bt*2+h)*256+i][256]
                        const unsigned short* __restrict__ WhT,  // [(bt*2+h)*128+f][256]
                        unsigned short* __restrict__ Hp) {       // [bt*256+i][h*128+f]
  int gw   = (blockIdx.x * blockDim.x + threadIdx.x) >> 5;  // 8192 waves
  int lane = threadIdx.x & 31;
  int ng = gw & 1;            // 2 groups of 64 f-columns
  int mt = (gw >> 1) & 15;    // i tile
  int bh = gw >> 5;           // 0..255
  int r = lane & 15, hi = lane >> 4;
  const unsigned short* arow  = Att + ((size_t)bh*NN + mt*16 + r) * NN;
  const unsigned short* brow0 = WhT + ((size_t)bh*HH + ng*64 + r) * NN;
  v8f acc[4] = {vzero8(), vzero8(), vzero8(), vzero8()};
#pragma unroll
  for (int kb = 0; kb < NN; kb += 32) {
    Frag a = load_a(arow, kb, hi);
#pragma unroll
    for (int q = 0; q < 4; ++q)
      acc[q] = wmma_bf16(a, load_b(brow0 + (size_t)q*16*NN, kb, hi), acc[q]);
  }
  int bt = bh >> 1, head = bh & 1;
#pragma unroll
  for (int q = 0; q < 4; ++q) {
    int f = ng*64 + q*16 + r;
#pragma unroll
    for (int v = 0; v < 8; ++v) {
      int i = mt*16 + v + 8*hi;
      Hp[(size_t)(bt*NN + i)*(HEADS*HH) + head*HH + f] = f2bf(acc[q][v]);
    }
  }
}

// ------------------------------------- hgat = elu(hp @ head_W^T)
// Register-blocked: 16(m) x 64(f) per wave.
__global__ void head_gemm(const unsigned short* __restrict__ Hp,
                          const unsigned short* __restrict__ Wo,  // [128][256]
                          unsigned short* __restrict__ Hg) {      // [bt*256+node][128]
  int gw   = (blockIdx.x * blockDim.x + threadIdx.x) >> 5;  // 4096 waves
  int lane = threadIdx.x & 31;
  int ng = gw & 1;
  int mt = gw >> 1;           // 2048
  int r = lane & 15, hi = lane >> 4;
  const unsigned short* arow  = Hp + (size_t)(mt*16 + r) * 256;
  const unsigned short* brow0 = Wo + (size_t)(ng*64 + r) * 256;
  v8f acc[4] = {vzero8(), vzero8(), vzero8(), vzero8()};
#pragma unroll
  for (int kb = 0; kb < 256; kb += 32) {
    Frag a = load_a(arow, kb, hi);
#pragma unroll
    for (int q = 0; q < 4; ++q)
      acc[q] = wmma_bf16(a, load_b(brow0 + (size_t)q*16*256, kb, hi), acc[q]);
  }
#pragma unroll
  for (int q = 0; q < 4; ++q) {
    int f = ng*64 + q*16 + r;
#pragma unroll
    for (int v = 0; v < 8; ++v) {
      int row = mt*16 + v + 8*hi;
      float x = acc[q][v];
      float e = (x > 0.f) ? x : (__expf(x) - 1.f);          // elu
      Hg[(size_t)row*HH + f] = f2bf(e);
    }
  }
}

// ----------------------------------------------------- one GRU step
// gi = x@Wih^T + bih, gh = h@Whh^T + bhh  (both via WMMA, K=128)
// h' = (1-z)*tanh(gi_n + r*gh_n) + z*h
// A fragments (x and h) are each reused across 3 WMMAs per k-step.
__global__ void gru_step(const unsigned short* __restrict__ Xbase, int mode, int t,
                         const unsigned short* __restrict__ Hpbf,
                         const float* __restrict__ Hpf,
                         const unsigned short* __restrict__ Wih,
                         const unsigned short* __restrict__ Whh,
                         const float* __restrict__ bih, const float* __restrict__ bhh,
                         float* __restrict__ Hnf, unsigned short* __restrict__ Hnbf,
                         unsigned short* __restrict__ Yout) {
  int gw   = (blockIdx.x * blockDim.x + threadIdx.x) >> 5;  // 3072
  int lane = threadIdx.x & 31;
  int nt = gw & 7;          // j tile (H/16)
  int mt = gw >> 3;         // 384 m tiles
  int r = lane & 15, hi = lane >> 4;
  int arowIdx = mt*16 + r;
  const unsigned short* xrow;
  if (mode == 0) {
    xrow = Xbase + (size_t)arowIdx * HH;               // contiguous layer-1 input
  } else {
    int si   = arowIdx % 3;                            // scale index
    int bn   = arowIdx / 3;
    int node = bn & 255, b = bn >> 8;
    int sc   = (si == 0) ? 1 : ((si == 1) ? 3 : 5);
    int tt2  = t * sc;
    xrow = (tt2 < TT) ? (Xbase + (size_t)((b*TT + tt2)*NN + node) * HH)
                      : (const unsigned short*)0;      // zero padding
  }
  const unsigned short* hrow = Hpbf + (size_t)arowIdx * HH;
  int j = nt*16 + r;
  const unsigned short* bri = Wih + (size_t)(0   + j)*HH;
  const unsigned short* bzi = Wih + (size_t)(128 + j)*HH;
  const unsigned short* bni = Wih + (size_t)(256 + j)*HH;
  const unsigned short* brh = Whh + (size_t)(0   + j)*HH;
  const unsigned short* bzh = Whh + (size_t)(128 + j)*HH;
  const unsigned short* bnh = Whh + (size_t)(256 + j)*HH;
  v8f gir = vzero8(), giz = vzero8(), gin = vzero8();
  v8f ghr = vzero8(), ghz = vzero8(), ghn = vzero8();
#pragma unroll
  for (int kb = 0; kb < HH; kb += 32) {
    Frag ax;
    if (xrow) ax = load_a(xrow, kb, hi);
    else { ax.u[0] = make_uint4(0,0,0,0); ax.u[1] = make_uint4(0,0,0,0); }
    Frag ah = load_a(hrow, kb, hi);
    gir = wmma_bf16(ax, load_b(bri, kb, hi), gir);
    giz = wmma_bf16(ax, load_b(bzi, kb, hi), giz);
    gin = wmma_bf16(ax, load_b(bni, kb, hi), gin);
    ghr = wmma_bf16(ah, load_b(brh, kb, hi), ghr);
    ghz = wmma_bf16(ah, load_b(bzh, kb, hi), ghz);
    ghn = wmma_bf16(ah, load_b(bnh, kb, hi), ghn);
  }
  float bir = bih[j],       bhr2 = bhh[j];
  float biz = bih[128 + j], bhz2 = bhh[128 + j];
  float bin = bih[256 + j], bhn2 = bhh[256 + j];
#pragma unroll
  for (int v = 0; v < 8; ++v) {
    int m = mt*16 + v + 8*hi;
    float hp = Hpf[(size_t)m*HH + j];
    float rg = 1.f / (1.f + __expf(-(gir[v] + ghr[v] + bir + bhr2)));
    float zg = 1.f / (1.f + __expf(-(giz[v] + ghz[v] + biz + bhz2)));
    float ng = tanhf(gin[v] + bin + rg*(ghn[v] + bhn2));
    float hn = (1.f - zg)*ng + zg*hp;
    Hnf[(size_t)m*HH + j]  = hn;
    unsigned short hb = f2bf(hn);
    Hnbf[(size_t)m*HH + j] = hb;
    if (Yout) Yout[(size_t)m*HH + j] = hb;
  }
}

// --------------------------------------- scale softmax + weighted sum
__global__ void final_mix(const float* __restrict__ hlast,   // [2048*3][128]
                          const float* __restrict__ sW,      // [3][384]
                          const float* __restrict__ sb,      // [3]
                          float* __restrict__ out) {         // [2048][128]
  __shared__ float red[3][128];
  __shared__ float wsm[3];
  int row = blockIdx.x;           // 0..2047 = b*256+node
  int tid = threadIdx.x;          // 128
  const float* last = hlast + (size_t)row*3*HH;
#pragma unroll
  for (int s = 0; s < 3; ++s) {
    float p = 0.f;
    for (int k = tid; k < 3*HH; k += 128) p += last[k]*sW[s*3*HH + k];
    red[s][tid] = p;
  }
  __syncthreads();
  if (tid == 0) {
    float l[3];
    for (int s = 0; s < 3; ++s) {
      float a = 0.f;
      for (int k = 0; k < 128; ++k) a += red[s][k];
      l[s] = a + sb[s];
    }
    float m = fmaxf(l[0], fmaxf(l[1], l[2]));
    float e0 = __expf(l[0]-m), e1 = __expf(l[1]-m), e2 = __expf(l[2]-m);
    float inv = 1.f / (e0 + e1 + e2);
    wsm[0] = e0*inv; wsm[1] = e1*inv; wsm[2] = e2*inv;
  }
  __syncthreads();
  out[(size_t)row*HH + tid] = last[0*HH + tid]*wsm[0]
                            + last[1*HH + tid]*wsm[1]
                            + last[2*HH + tid]*wsm[2];
}

extern "C" void kernel_launch(void* const* d_in, const int* in_sizes, int n_in,
                              void* d_out, int out_size, void* d_ws, size_t ws_size,
                              hipStream_t stream) {
  const float* x      = (const float*)d_in[0];
  const float* proj   = (const float*)d_in[1];
  const float* sim    = (const float*)d_in[2];
  const int*   quad   = (const int*)  d_in[3];
  const float* W_gat  = (const float*)d_in[4];
  const float* a_attn = (const float*)d_in[5];
  const float* w_proj = (const float*)d_in[6];
  const float* w_sim  = (const float*)d_in[7];
  const float* qb     = (const float*)d_in[8];
  const float* head_W = (const float*)d_in[9];
  const float* Wih0   = (const float*)d_in[10];
  const float* Whh0   = (const float*)d_in[11];
  const float* bih0   = (const float*)d_in[12];
  const float* bhh0   = (const float*)d_in[13];
  const float* Wih1   = (const float*)d_in[14];
  const float* Whh1   = (const float*)d_in[15];
  const float* bih1   = (const float*)d_in[16];
  const float* bhh1   = (const float*)d_in[17];
  const float* sW     = (const float*)d_in[18];
  const float* sb     = (const float*)d_in[19];
  (void)in_sizes; (void)n_in; (void)out_size; (void)ws_size;

  char* ws = (char*)d_ws;
  size_t off = 0;
  auto alloc = [&](size_t bytes) -> char* {
    char* p = ws + off;
    off = (off + bytes + 255) & ~(size_t)255;
    return p;
  };
  unsigned short* xbf    = (unsigned short*)alloc((size_t)ROWS_X*FIN*2);
  unsigned short* wgatbf = (unsigned short*)alloc(256*128*2);
  unsigned short* howbf  = (unsigned short*)alloc(128*256*2);
  unsigned short* wih0bf = (unsigned short*)alloc(384*128*2);
  unsigned short* whh0bf = (unsigned short*)alloc(384*128*2);
  unsigned short* wih1bf = (unsigned short*)alloc(384*128*2);
  unsigned short* whh1bf = (unsigned short*)alloc(384*128*2);
  float*          Whf    = (float*)         alloc((size_t)BT*HEADS*NN*HH*4);
  unsigned short* WhT    = (unsigned short*)alloc((size_t)BT*HEADS*HH*NN*2);
  float*          s1     = (float*)         alloc((size_t)65536*4);
  float*          s2     = (float*)         alloc((size_t)65536*4);
  float*          spd    = (float*)         alloc((size_t)65536*4);
  float*          ssd    = (float*)         alloc((size_t)65536*4);
  unsigned short* attbf  = (unsigned short*)alloc((size_t)BT*HEADS*NN*NN*2);
  unsigned short* hpbf   = (unsigned short*)alloc((size_t)BT*NN*256*2);
  unsigned short* hgbf   = (unsigned short*)alloc((size_t)ROWS_X*HH*2);
  unsigned short* y0bf   = (unsigned short*)alloc((size_t)TT*SEQ*HH*2);
  float*          hAf    = (float*)         alloc((size_t)SEQ*HH*4);
  float*          hBf    = (float*)         alloc((size_t)SEQ*HH*4);
  unsigned short* hAbf   = (unsigned short*)alloc((size_t)SEQ*HH*2);
  unsigned short* hBbf   = (unsigned short*)alloc((size_t)SEQ*HH*2);

  // ---- bf16 conversions
  cvt_bf16<<<(ROWS_X*FIN + 255)/256, 256, 0, stream>>>(x, xbf, ROWS_X*FIN);
  cvt_bf16<<<(32768 + 255)/256, 256, 0, stream>>>(W_gat,  wgatbf, 32768);
  cvt_bf16<<<(32768 + 255)/256, 256, 0, stream>>>(head_W, howbf,  32768);
  cvt_bf16<<<(49152 + 255)/256, 256, 0, stream>>>(Wih0, wih0bf, 49152);
  cvt_bf16<<<(49152 + 255)/256, 256, 0, stream>>>(Whh0, whh0bf, 49152);
  cvt_bf16<<<(49152 + 255)/256, 256, 0, stream>>>(Wih1, wih1bf, 49152);
  cvt_bf16<<<(49152 + 255)/256, 256, 0, stream>>>(Whh1, whh1bf, 49152);

  // ---- GAT phase
  wh_gemm  <<<1024, 256, 0, stream>>>(xbf, wgatbf, Whf, WhT);          // 8192 waves
  row_dots <<<8192, 256, 0, stream>>>(Whf, a_attn, w_proj, w_sim, s1, s2, spd, ssd);
  attn_topk<<<16384, 128, 0, stream>>>(s1, s2, spd, ssd, proj, sim, quad, qb, attbf);
  hp_gemm  <<<1024, 256, 0, stream>>>(attbf, WhT, hpbf);               // 8192 waves
  head_gemm<<<512, 256, 0, stream>>>(hpbf, howbf, hgbf);               // 4096 waves

  // ---- GRU layer 0 (strided multi-scale input from hgat)
  hipMemsetAsync(hAf,  0, (size_t)SEQ*HH*4, stream);
  hipMemsetAsync(hAbf, 0, (size_t)SEQ*HH*2, stream);
  float* pf = hAf;  float* nf2 = hBf;
  unsigned short* pbf = hAbf; unsigned short* nbf = hBbf;
  for (int t = 0; t < TT; ++t) {
    gru_step<<<384, 256, 0, stream>>>(hgbf, 1, t, pbf, pf, wih0bf, whh0bf,
                                      bih0, bhh0, nf2, nbf,
                                      y0bf + (size_t)t*SEQ*HH);
    float* tf = pf; pf = nf2; nf2 = tf;
    unsigned short* tb = pbf; pbf = nbf; nbf = tb;
  }

  // ---- GRU layer 1 (contiguous input from y0)
  hipMemsetAsync(pf,  0, (size_t)SEQ*HH*4, stream);
  hipMemsetAsync(pbf, 0, (size_t)SEQ*HH*2, stream);
  for (int t = 0; t < TT; ++t) {
    gru_step<<<384, 256, 0, stream>>>(y0bf + (size_t)t*SEQ*HH, 0, t, pbf, pf,
                                      wih1bf, whh1bf, bih1, bhh1, nf2, nbf,
                                      (unsigned short*)nullptr);
    float* tf = pf; pf = nf2; nf2 = tf;
    unsigned short* tb = pbf; pbf = nbf; nbf = tb;
  }

  // ---- scale mixing -> output (8,256,128)
  final_mix<<<2048, 128, 0, stream>>>(pf, sW, sb, (float*)d_out);
}
